// SampleRateNetwork_91061896609826
// MI455X (gfx1250) — matrix operand order repaired and checked by reading, not compiled, not run on hardware
//
#include <hip/hip_runtime.h>

#define B_ROWS 65536
#define ROWTILES (B_ROWS / 16)
#define USE_ASYNC_LDS 1

typedef __attribute__((ext_vector_type(16))) __bf16 bf16x16;
typedef __attribute__((ext_vector_type(8)))  float  floatx8;
typedef __attribute__((ext_vector_type(4)))  unsigned int uintx4;

union FragU { uintx4 u[2]; bf16x16 v; };

__device__ __forceinline__ unsigned short f2bf(float f) {
    unsigned int u = __float_as_uint(f);
    unsigned int r = u + 0x7FFFu + ((u >> 16) & 1u);   // round-to-nearest-even
    return (unsigned short)(r >> 16);
}

__device__ __forceinline__ float sigm_f(float x) {
    return 1.0f / (1.0f + __expf(-x));
}
__device__ __forceinline__ float tanh_f(float x) {
    x = fminf(fmaxf(x, -15.0f), 15.0f);
    float e = __expf(2.0f * x);
    return (e - 1.0f) / (e + 1.0f);
}

// A-fragment (16x32 bf16) from a row-major bf16 matrix.
// Contiguous order: lanes 0-15 hold K k0..k0+15; lanes 16-31 hold k0+16..k0+31.
__device__ __forceinline__ bf16x16 load_afrag(const unsigned short* __restrict__ rowptr,
                                              int k0, int hi) {
    const unsigned short* p = rowptr + k0 + hi * 16;
    FragU f;
    f.u[0] = *(const uintx4*)(p);
    f.u[1] = *(const uintx4*)(p + 8);
    return f.v;
}

// B-fragment (32x16 bf16) from pre-packed lane-contiguous storage (global).
__device__ __forceinline__ bf16x16 load_bfrag(const unsigned short* __restrict__ fragbase,
                                              int lane) {
    const unsigned short* p = fragbase + lane * 16;
    FragU f;
    f.u[0] = *(const uintx4*)(p);
    f.u[1] = *(const uintx4*)(p + 8);
    return f.v;
}

// ---------------------------------------------------------------------------
// Pack fp32 row-major weight [K x N] into bf16 B-fragment layout.
// Fragment f = nt*NKT + kt. Element lane*16 + kk <- W[kt*32 + (lane>>4)*16 + kk, nt*16 + (lane&15)].
// Rows k >= K are zero padded (for the K=16 FC weights padded to K=32).
__global__ void pack_b_kernel(const float* __restrict__ W, unsigned short* __restrict__ dst,
                              int K, int N, int NKT, int total) {
    int idx = blockIdx.x * blockDim.x + threadIdx.x;
    if (idx >= total) return;
    int f    = idx >> 9;
    int w    = idx & 511;
    int lane = w >> 4;
    int kk   = w & 15;
    int nt   = f / NKT;
    int kt   = f % NKT;
    int k = kt * 32 + (lane >> 4) * 16 + kk;
    int n = nt * 16 + (lane & 15);
    float v = (k < K) ? W[k * N + n] : 0.0f;
    dst[idx] = f2bf(v);
}

// ---------------------------------------------------------------------------
// Concatenate x1[B,384] | x2[B,128] -> e[B,512] in packed bf16 (one dword = 2 elems).
__global__ void concat_bf16_kernel(const float* __restrict__ x1, const float* __restrict__ x2,
                                   unsigned int* __restrict__ e, int total) {
    int idx = blockIdx.x * blockDim.x + threadIdx.x;
    if (idx >= total) return;
    int row = idx >> 8;          // 256 dwords per row
    int col = (idx & 255) * 2;
    float a, b;
    if (col < 384) {
        const float* p = x1 + row * 384 + col;
        a = p[0]; b = p[1];
    } else {
        const float* p = x2 + row * 128 + (col - 384);
        a = p[0]; b = p[1];
    }
    e[idx] = (unsigned int)f2bf(a) | ((unsigned int)f2bf(b) << 16);
}

// ---------------------------------------------------------------------------
// GEMM1 (65536x512 @ 512x1152) fused with GRU1 -> h1 bf16 [B,384].
// Block = 8 waves; block owns 32 row-tiles x 1 u-column tile.
// The 48KB B slab (3 gates x 16 k-tiles x 1KB) is staged to LDS once via
// async global->LDS loads, then every wave computes 4 row-tiles (12 accums).
__global__ void __launch_bounds__(256)
gemm1_gru1_kernel(const unsigned short* __restrict__ eB,
                  const unsigned short* __restrict__ Wg1,   // packed, NKT=16, 72 n-tiles
                  const float* __restrict__ bias,           // [2,1152] flat
                  unsigned short* __restrict__ h1) {
    __shared__ unsigned short smem[3 * 8192];   // 48 KB

    int tid  = threadIdx.x;
    int lane = tid & 31;
    int wid  = tid >> 5;
    int ut    = blockIdx.x % 24;       // u-column tile (0..23)
    int group = blockIdx.x / 24;       // 32-row-tile group (0..127)

    // ---- stage B slab: gate g slab = 16KB contiguous in packed layout ----
#pragma unroll
    for (int g = 0; g < 3; ++g) {
        const unsigned short* gs = Wg1 + (size_t)(ut + 24 * g) * 16 * 512;
#pragma unroll
        for (int i = 0; i < 4; ++i) {
            int e = (tid + i * 256) * 8;            // 8 bf16 = 16 bytes per thread-chunk
#if USE_ASYNC_LDS
            // LDS byte offset = low 32 bits of the generic address (ISA 10.2);
            // the ptrtoint also makes `smem` escape so the "memory" clobber
            // keeps the later ds_load_b128 reads live.
            unsigned lds_off = (unsigned)(size_t)(&smem[g * 8192 + e]);
            const void* gp = (const void*)(gs + e);
            asm volatile("global_load_async_to_lds_b128 %0, %1, off"
                         :: "v"(lds_off), "v"(gp) : "memory");
#else
            *(uintx4*)(&smem[g * 8192 + e]) = *(const uintx4*)(gs + e);
#endif
        }
    }
#if USE_ASYNC_LDS
    asm volatile("s_wait_asynccnt 0x0" ::: "memory");
#endif
    __syncthreads();

    int hi = lane >> 4;
    int ln = lane & 15;
    int rt0 = group * 32 + wid * 4;

    const unsigned short* arow0 = eB + (size_t)((rt0 + 0) * 16 + ln) * 512;
    const unsigned short* arow1 = eB + (size_t)((rt0 + 1) * 16 + ln) * 512;
    const unsigned short* arow2 = eB + (size_t)((rt0 + 2) * 16 + ln) * 512;
    const unsigned short* arow3 = eB + (size_t)((rt0 + 3) * 16 + ln) * 512;

    floatx8 acc[4][3];
#pragma unroll
    for (int j = 0; j < 4; ++j)
#pragma unroll
        for (int g = 0; g < 3; ++g)
            acc[j][g] = (floatx8){};

#pragma unroll 4
    for (int kt = 0; kt < 16; ++kt) {
        // B fragments from LDS (shared by all 8 waves)
        bf16x16 b[3];
#pragma unroll
        for (int g = 0; g < 3; ++g) {
            int idx = g * 8192 + kt * 512 + lane * 16;
            FragU f;
            f.u[0] = *(const uintx4*)(&smem[idx]);
            f.u[1] = *(const uintx4*)(&smem[idx + 8]);
            b[g] = f.v;
        }
        bf16x16 a0 = load_afrag(arow0, kt * 32, hi);
        bf16x16 a1 = load_afrag(arow1, kt * 32, hi);
        bf16x16 a2 = load_afrag(arow2, kt * 32, hi);
        bf16x16 a3 = load_afrag(arow3, kt * 32, hi);
#pragma unroll
        for (int g = 0; g < 3; ++g) {
            acc[0][g] = __builtin_amdgcn_wmma_f32_16x16x32_bf16(false, a0, false, b[g], (short)0, acc[0][g], false, false);
            acc[1][g] = __builtin_amdgcn_wmma_f32_16x16x32_bf16(false, a1, false, b[g], (short)0, acc[1][g], false, false);
            acc[2][g] = __builtin_amdgcn_wmma_f32_16x16x32_bf16(false, a2, false, b[g], (short)0, acc[2][g], false, false);
            acc[3][g] = __builtin_amdgcn_wmma_f32_16x16x32_bf16(false, a3, false, b[g], (short)0, acc[3][g], false, false);
        }
    }

    int n = ut * 16 + ln;
    float b0z = bias[n],       b1z = bias[1152 + n];
    float b0r = bias[384 + n], b1r = bias[1152 + 384 + n];
    float b0h = bias[768 + n], b1h = bias[1152 + 768 + n];

#pragma unroll
    for (int j = 0; j < 4; ++j) {
#pragma unroll
        for (int v = 0; v < 8; ++v) {
            int m = (rt0 + j) * 16 + hi * 8 + v;
            float z  = sigm_f(acc[j][0][v] + b0z + b1z);
            float r  = sigm_f(acc[j][1][v] + b0r + b1r);
            float hc = tanh_f(acc[j][2][v] + b0h + r * b1h);
            h1[(size_t)m * 384 + n] = f2bf((1.0f - z) * hc);
        }
    }
}

// ---------------------------------------------------------------------------
// GEMM2 (65536x384 @ 384x48) fused with GRU2 -> h2 bf16 [B,32] (cols 16..31 zero pad).
__global__ void __launch_bounds__(256)
gemm2_gru2_kernel(const unsigned short* __restrict__ h1,
                  const unsigned short* __restrict__ Wg2,   // packed, NKT=12, 3 n-tiles
                  const float* __restrict__ bias,           // [2,48] flat
                  unsigned short* __restrict__ h2) {
    int lane = threadIdx.x & 31;
    int wid  = threadIdx.x >> 5;
    int rowTile = blockIdx.x * 8 + wid;
    int hi = lane >> 4;
    int ln = lane & 15;

    const unsigned short* arow = h1 + (size_t)(rowTile * 16 + ln) * 384;

    floatx8 az = {}, ar = {}, ah = {};
    for (int kt = 0; kt < 12; ++kt) {
        bf16x16 a  = load_afrag(arow, kt * 32, hi);
        bf16x16 bz = load_bfrag(Wg2 + ((0 * 12) + kt) * 512, lane);
        bf16x16 br = load_bfrag(Wg2 + ((1 * 12) + kt) * 512, lane);
        bf16x16 bh = load_bfrag(Wg2 + ((2 * 12) + kt) * 512, lane);
        az = __builtin_amdgcn_wmma_f32_16x16x32_bf16(false, a, false, bz, (short)0, az, false, false);
        ar = __builtin_amdgcn_wmma_f32_16x16x32_bf16(false, a, false, br, (short)0, ar, false, false);
        ah = __builtin_amdgcn_wmma_f32_16x16x32_bf16(false, a, false, bh, (short)0, ah, false, false);
    }

    int n = ln;
    float b0z = bias[n],      b1z = bias[48 + n];
    float b0r = bias[16 + n], b1r = bias[48 + 16 + n];
    float b0h = bias[32 + n], b1h = bias[48 + 32 + n];

#pragma unroll
    for (int v = 0; v < 8; ++v) {
        int m = rowTile * 16 + hi * 8 + v;
        float z  = sigm_f(az[v] + b0z + b1z);
        float r  = sigm_f(ar[v] + b0r + b1r);
        float hc = tanh_f(ah[v] + b0h + r * b1h);
        h2[m * 32 + n]      = f2bf((1.0f - z) * hc);
        h2[m * 32 + 16 + n] = 0;                       // K pad for next GEMM
    }
}

// ---------------------------------------------------------------------------
// FC pair (65536x16 @ 16x256, K padded to 32) + tanh/gain combine + softmax(256).
__global__ void __launch_bounds__(256)
fc_softmax_kernel(const unsigned short* __restrict__ h2,
                  const unsigned short* __restrict__ Wf1,   // packed, NKT=1, 16 n-tiles
                  const unsigned short* __restrict__ Wf2,
                  const float* __restrict__ fb1, const float* __restrict__ fb2,
                  const float* __restrict__ fg1, const float* __restrict__ fg2,
                  float* __restrict__ out) {
    int lane = threadIdx.x & 31;
    int wid  = threadIdx.x >> 5;
    int rowTile = blockIdx.x * 8 + wid;
    int hi = lane >> 4;
    int ln = lane & 15;

    const unsigned short* arow = h2 + (size_t)(rowTile * 16 + ln) * 32;
    bf16x16 a = load_afrag(arow, 0, hi);

    float d[16][8];
#pragma unroll
    for (int t = 0; t < 16; ++t) {
        floatx8 c1 = {}, c2 = {};
        bf16x16 b1f = load_bfrag(Wf1 + t * 512, lane);
        bf16x16 b2f = load_bfrag(Wf2 + t * 512, lane);
        c1 = __builtin_amdgcn_wmma_f32_16x16x32_bf16(false, a, false, b1f, (short)0, c1, false, false);
        c2 = __builtin_amdgcn_wmma_f32_16x16x32_bf16(false, a, false, b2f, (short)0, c2, false, false);
        int c = t * 16 + ln;
        float g1 = fg1[c], bb1 = fb1[c];
        float g2 = fg2[c], bb2 = fb2[c];
#pragma unroll
        for (int v = 0; v < 8; ++v)
            d[t][v] = g1 * tanh_f(c1[v] + bb1) + g2 * tanh_f(c2[v] + bb2);
    }

#pragma unroll
    for (int v = 0; v < 8; ++v) {
        float rm = d[0][v];
#pragma unroll
        for (int t = 1; t < 16; ++t) rm = fmaxf(rm, d[t][v]);
        rm = fmaxf(rm, __shfl_xor(rm, 1, 32));
        rm = fmaxf(rm, __shfl_xor(rm, 2, 32));
        rm = fmaxf(rm, __shfl_xor(rm, 4, 32));
        rm = fmaxf(rm, __shfl_xor(rm, 8, 32));

        float s = 0.0f;
#pragma unroll
        for (int t = 0; t < 16; ++t) {
            float e = __expf(d[t][v] - rm);
            d[t][v] = e;
            s += e;
        }
        s += __shfl_xor(s, 1, 32);
        s += __shfl_xor(s, 2, 32);
        s += __shfl_xor(s, 4, 32);
        s += __shfl_xor(s, 8, 32);
        float inv = 1.0f / s;

        int m = rowTile * 16 + hi * 8 + v;
#pragma unroll
        for (int t = 0; t < 16; ++t)
            out[(size_t)m * 256 + t * 16 + ln] = d[t][v] * inv;
    }
}

// ---------------------------------------------------------------------------
extern "C" void kernel_launch(void* const* d_in, const int* in_sizes, int n_in,
                              void* d_out, int out_size, void* d_ws, size_t ws_size,
                              hipStream_t stream) {
    const float* x1   = (const float*)d_in[0];
    const float* x2   = (const float*)d_in[1];
    const float* g1k  = (const float*)d_in[2];   // [512,1152]
    const float* g1b  = (const float*)d_in[4];   // [2,1152]  (rkernel d_in[3] dead: h0==0)
    const float* g2k  = (const float*)d_in[5];   // [384,48]
    const float* g2b  = (const float*)d_in[7];   // [2,48]    (rkernel d_in[6] dead)
    const float* fw1  = (const float*)d_in[8];   // [16,256]
    const float* fb1  = (const float*)d_in[9];
    const float* fw2  = (const float*)d_in[10];
    const float* fb2  = (const float*)d_in[11];
    const float* fg1  = (const float*)d_in[12];
    const float* fg2  = (const float*)d_in[13];
    float* out = (float*)d_out;

    char* ws = (char*)d_ws;
    size_t off_e   = 0;                                   // B*512 bf16 = 64 MiB
    size_t off_h1  = off_e  + (size_t)B_ROWS * 512 * 2;   // B*384 bf16 = 48 MiB
    size_t off_h2  = off_h1 + (size_t)B_ROWS * 384 * 2;   // B*32  bf16 =  4 MiB
    size_t off_w1  = off_h2 + (size_t)B_ROWS * 32 * 2;
    size_t off_w2  = off_w1 + (size_t)(72 * 16 * 512) * 2;
    size_t off_wf1 = off_w2 + (size_t)(3 * 12 * 512) * 2;
    size_t off_wf2 = off_wf1 + (size_t)(16 * 512) * 2;

    unsigned short* eB  = (unsigned short*)(ws + off_e);
    unsigned short* h1  = (unsigned short*)(ws + off_h1);
    unsigned short* h2  = (unsigned short*)(ws + off_h2);
    unsigned short* Wg1 = (unsigned short*)(ws + off_w1);
    unsigned short* Wg2 = (unsigned short*)(ws + off_w2);
    unsigned short* Wf1 = (unsigned short*)(ws + off_wf1);
    unsigned short* Wf2 = (unsigned short*)(ws + off_wf2);

    // Weight packing
    {
        int tot = 72 * 16 * 512;
        pack_b_kernel<<<(tot + 255) / 256, 256, 0, stream>>>(g1k, Wg1, 512, 1152, 16, tot);
    }
    {
        int tot = 3 * 12 * 512;
        pack_b_kernel<<<(tot + 255) / 256, 256, 0, stream>>>(g2k, Wg2, 384, 48, 12, tot);
    }
    {
        int tot = 16 * 512;
        pack_b_kernel<<<(tot + 255) / 256, 256, 0, stream>>>(fw1, Wf1, 16, 256, 1, tot);
        pack_b_kernel<<<(tot + 255) / 256, 256, 0, stream>>>(fw2, Wf2, 16, 256, 1, tot);
    }

    // Concat + fp32 -> bf16
    {
        int tot = B_ROWS * 256;   // dwords
        concat_bf16_kernel<<<tot / 256, 256, 0, stream>>>(x1, x2, (unsigned int*)eB, tot);
    }

    // GEMM1 + GRU1: 128 row-groups (32 rowtiles each) x 24 u-tiles
    gemm1_gru1_kernel<<<(ROWTILES / 32) * 24, 256, 0, stream>>>(eB, Wg1, g1b, h1);

    // GEMM2 + GRU2
    gemm2_gru2_kernel<<<ROWTILES / 8, 256, 0, stream>>>(h1, Wg2, g2b, h2);

    // FC pair + softmax
    fc_softmax_kernel<<<ROWTILES / 8, 256, 0, stream>>>(h2, Wf1, Wf2, fb1, fb2, fg1, fg2, out);

    (void)in_sizes; (void)n_in; (void)out_size; (void)ws_size;
}
